// DeepSets_46256797778106
// MI455X (gfx1250) — matrix operand order repaired and checked by reading, not compiled, not run on hardware
//
#include <hip/hip_runtime.h>
#include <math.h>

// ---------- plain ext-vector types ----------
typedef __attribute__((ext_vector_type(16))) __bf16    v16bf;
typedef __attribute__((ext_vector_type(8)))  float     v8f;
typedef __attribute__((ext_vector_type(2)))  float     f32x2;
typedef __attribute__((ext_vector_type(4)))  float     f32x4;
typedef __attribute__((ext_vector_type(4)))  unsigned  u32x4;

union FragBF {            // 32 bytes = 8 VGPRs/lane = one 16x32 bf16 A or 32x16 bf16 B fragment
    u32x4 q[2];
    v16bf v;
};

// fp32 -> bf16, round-to-nearest-even
__device__ __forceinline__ unsigned f2bf(float f) {
    unsigned u = __float_as_uint(f);
    u += 0x7fffu + ((u >> 16) & 1u);
    return u >> 16;
}
__device__ __forceinline__ unsigned packbf2(float a, float b) {
    return f2bf(a) | (f2bf(b) << 16);
}
__device__ __forceinline__ u32x4 pack8(f32x4 a, f32x4 b) {
    u32x4 r;
    r.x = packbf2(a.x, a.y);
    r.y = packbf2(a.z, a.w);
    r.z = packbf2(b.x, b.y);
    r.w = packbf2(b.z, b.w);
    return r;
}

// Native CDNA5 transcendental tanh (V_TANH_F32); asm fallback if builtin missing.
__device__ __forceinline__ float fast_tanh(float x) {
#if __has_builtin(__builtin_amdgcn_tanhf)
    return __builtin_amdgcn_tanhf(x);
#else
    float r;
    asm volatile("v_tanh_f32 %0, %1\n\tv_nop" : "=v"(r) : "v"(x));
    return r;
#endif
}

#define NSEG 4096
#define H    128
#define TWOH 256
#define W1PITCH 136   // bf16 elements per LDS row (272B pitch -> conflict-free b128 reads)

// ---------------- kernel 0a: convert W1 fp32 -> bf16 in workspace ----------------
__global__ void cvt_w1_kernel(const float* __restrict__ w, unsigned short* __restrict__ o, int n) {
    int i = blockIdx.x * blockDim.x + threadIdx.x;
    if (i < n) o[i] = (unsigned short)f2bf(w[i]);
}

// ---------------- kernel 0b: zero the segment-sum accumulator ----------------
__global__ void zero_f32_kernel(float* __restrict__ p, int n) {
    int i = blockIdx.x * blockDim.x + threadIdx.x;
    if (i < n) p[i] = 0.0f;
}

// ---------------- kernel 1: fused fc1 (bf16 WMMA) + tanh + sorted segment-sum ----------------
// grid = N/128 blocks, 256 threads (8 waves). Block tile: 128 rows x 256 cols, K = 128.
__global__ __launch_bounds__(256) void fc1_pool_kernel(
    const float* __restrict__ x,            // [N,128] fp32
    const long long* __restrict__ batch,    // [N] int64, sorted
    const unsigned short* __restrict__ w1bf,// [256,128] bf16
    const float* __restrict__ b1,           // [256]
    float* __restrict__ sumx)               // [4096,256] fp32 (atomic accum)
{
    __shared__ unsigned short w1s[128 * W1PITCH]; // one half of W1 (128 out-channels)
    __shared__ float b1s[TWOH];
    __shared__ int   segs[128];

    const int tid  = threadIdx.x;
    const int lane = tid & 31;
    const int wave = tid >> 5;
    const int lr   = lane & 15;   // lane row/col within 16
    const int half = lane >> 4;   // 0: lanes 0-15, 1: lanes 16-31
    const long long rowbase = (long long)blockIdx.x * 128;

    b1s[tid] = b1[tid];
    if (tid < 128) segs[tid] = (int)batch[rowbase + tid];

    // ---- A fragments: 16 rows per wave, full K=128, fp32->bf16 in registers ----
    // A layout (16-bit 16x32): lane<16 holds K {k..k+7} and {k+16..k+23}; lane>=16 the +8 chunks.
    FragBF a[4];
    {
        const float* xr = x + (rowbase + wave * 16 + lr) * H;
        #pragma unroll
        for (int k4 = 0; k4 < 4; ++k4) {
            const float* p0 = xr + k4 * 32 + half * 8;
            f32x4 c0 = *(const f32x4*)(p0);
            f32x4 c1 = *(const f32x4*)(p0 + 4);
            f32x4 c2 = *(const f32x4*)(p0 + 16);
            f32x4 c3 = *(const f32x4*)(p0 + 20);
            a[k4].q[0] = pack8(c0, c1);
            a[k4].q[1] = pack8(c2, c3);
        }
    }

    __syncthreads();  // segs visible to all waves

    const int mbase = wave * 16 + half * 8;   // 8 CONSECUTIVE rows of this lane's D column

    // Preload this lane's 8 segment ids into registers (no LDS in the epilogue).
    int segr[8];
    #pragma unroll
    for (int r = 0; r < 8; ++r) segr[r] = segs[mbase + r];

    // Wave-uniform check, forced scalar via readfirstlane: all lanes read the same
    // two segs entries, so this is a pure s_cmp/s_cbranch -- no exec-mask dance.
    // (avg segment = 256 rows -> uniform ~94% of the time)
    const bool seg_uniform =
        __builtin_amdgcn_readfirstlane((int)(segs[wave * 16] == segs[wave * 16 + 15])) != 0;

    // epilogue: bias + native tanh + sorted-segment merge + atomic scatter
    auto epilogue = [&](v8f acc, int nt) {
        const int   ocol = nt * 16 + lr;
        const float bias = b1s[ocol];
        if (seg_uniform) {
            // fast path: all 8 rows in one segment -> one unconditional atomic
            float v = fast_tanh(acc[0] + bias);
            #pragma unroll
            for (int r = 1; r < 8; ++r) v += fast_tanh(acc[r] + bias);
            atomicAdd(&sumx[segr[0] * TWOH + ocol], v);
        } else {
            // slow path: running merge over consecutive rows, flush on boundary
            float vcur = fast_tanh(acc[0] + bias);
            int   scur = segr[0];
            #pragma unroll
            for (int r = 1; r < 8; ++r) {
                float v = fast_tanh(acc[r] + bias);
                int   s = segr[r];
                if (s == scur) {
                    vcur += v;
                } else {
                    atomicAdd(&sumx[scur * TWOH + ocol], vcur);
                    scur = s;
                    vcur = v;
                }
            }
            atomicAdd(&sumx[scur * TWOH + ocol], vcur);
        }
    };

    const u32x4* wg = (const u32x4*)w1bf;   // 16 bf16 chunks of 8 per W1 row

    #pragma unroll 1
    for (int hb = 0; hb < 2; ++hb) {        // two halves of the 256 output channels
        __syncthreads();                    // prior half fully consumed
        #pragma unroll
        for (int i = 0; i < 8; ++i) {       // stage 128x128 bf16 of W1 into LDS (coalesced b128)
            int idx = i * 256 + tid;        // 0..2047
            int r  = idx >> 4;
            int ch = idx & 15;
            *(u32x4*)&w1s[r * W1PITCH + ch * 8] = wg[(hb * 128 + r) * 16 + ch];
        }
        __syncthreads();

        // Two independent accumulator chains per iteration: fills the WMMA->WMMA
        // D->C hazard window with the other chain's WMMA/ds work.
        #pragma unroll 1
        for (int ntl = 0; ntl < 8; ntl += 2) {
            const int lrow0 = (ntl + 0) * 16 + lr;   // rows within staged W1 half
            const int lrow1 = (ntl + 1) * 16 + lr;

            v8f acc0 = {0.f, 0.f, 0.f, 0.f, 0.f, 0.f, 0.f, 0.f};
            v8f acc1 = {0.f, 0.f, 0.f, 0.f, 0.f, 0.f, 0.f, 0.f};
            #pragma unroll
            for (int k4 = 0; k4 < 4; ++k4) {
                // B layout (16-bit 32x16): lane holds column `lr`, K = half*8..+7 and 16+half*8..+7 (matches A)
                FragBF b0, b1f;
                const unsigned short* p0 = &w1s[lrow0 * W1PITCH + k4 * 32 + half * 8];
                const unsigned short* p1 = &w1s[lrow1 * W1PITCH + k4 * 32 + half * 8];
                b0.q[0]  = *(const u32x4*)(p0);
                b0.q[1]  = *(const u32x4*)(p0 + 16);
                b1f.q[0] = *(const u32x4*)(p1);
                b1f.q[1] = *(const u32x4*)(p1 + 16);
                acc0 = __builtin_amdgcn_wmma_f32_16x16x32_bf16(
                           false, a[k4].v, false, b0.v,  (short)0, acc0, false, false);
                acc1 = __builtin_amdgcn_wmma_f32_16x16x32_bf16(
                           false, a[k4].v, false, b1f.v, (short)0, acc1, false, false);
            }
            epilogue(acc0, hb * 8 + ntl + 0);
            epilogue(acc1, hb * 8 + ntl + 1);
        }
    }
}

// ---------------- kernel 2: fc2 with fp32 WMMA (16x16x4_f32), K = 256 ----------------
// grid = 4096/16 = 256 blocks, 8 waves; wave w handles output-column tile w (128 cols / 16).
__global__ __launch_bounds__(256) void fc2_kernel(
    const float* __restrict__ sumx,   // [4096,256]
    const float* __restrict__ W2,     // [128,256] row-major (y[g,h] = sum_o sumx[g,o]*W2[h,o])
    const float* __restrict__ b2,     // [128]
    float* __restrict__ out)          // [4096,128]
{
    const int tid  = threadIdx.x;
    const int lane = tid & 31;
    const int wave = tid >> 5;
    const int lr   = lane & 15;
    const int half = lane >> 4;
    const int g0    = blockIdx.x * 16;
    const int hbase = wave * 16;

    // A (f32 16x4): lane row = lr, K = half*2 + {0,1}  -> contiguous float2
    // B (f32 4x16): lane col = lr, K = half*2 + {0,1}  -> contiguous float2 from W2 row
    const float* ap = sumx + (g0 + lr) * TWOH + half * 2;
    const float* bp = W2   + (hbase + lr) * TWOH + half * 2;

    v8f acc = {0.f, 0.f, 0.f, 0.f, 0.f, 0.f, 0.f, 0.f};
    #pragma unroll 4
    for (int k = 0; k < TWOH; k += 4) {
        f32x2 av = *(const f32x2*)(ap + k);
        f32x2 bv = *(const f32x2*)(bp + k);
        acc = __builtin_amdgcn_wmma_f32_16x16x4_f32(
                  false, av, false, bv, (short)0, acc, false, false);
    }

    const float bb = b2[hbase + lr];
    #pragma unroll
    for (int r = 0; r < 8; ++r) {
        int g = g0 + half * 8 + r;
        out[g * H + hbase + lr] = acc[r] + bb;
    }
}

// ---------------- host launch ----------------
extern "C" void kernel_launch(void* const* d_in, const int* in_sizes, int n_in,
                              void* d_out, int out_size, void* d_ws, size_t ws_size,
                              hipStream_t stream) {
    const float*     x     = (const float*)d_in[0];
    const long long* batch = (const long long*)d_in[1];
    const float*     W1    = (const float*)d_in[2];
    const float*     b1    = (const float*)d_in[3];
    const float*     W2    = (const float*)d_in[4];
    const float*     b2    = (const float*)d_in[5];
    float* out = (float*)d_out;

    const int N = in_sizes[0] / H;                 // 1048576 rows

    // workspace: [0,64KB) bf16 W1 ; [64KB, 64KB+4MB) fp32 segment sums
    unsigned short* w1bf = (unsigned short*)d_ws;
    float* sumx = (float*)((char*)d_ws + 65536);

    cvt_w1_kernel  <<<(TWOH * H + 255) / 256, 256, 0, stream>>>(W1, w1bf, TWOH * H);
    zero_f32_kernel<<<(NSEG * TWOH + 255) / 256, 256, 0, stream>>>(sumx, NSEG * TWOH);
    fc1_pool_kernel<<<N / 128, 256, 0, stream>>>(x, batch, w1bf, b1, sumx);
    fc2_kernel     <<<NSEG / 16, 256, 0, stream>>>(sumx, W2, b2, out);
}
